// VisionTransformer_87952340287952
// MI455X (gfx1250) — matrix operand order repaired and checked by reading, not compiled, not run on hardware
//
#include <hip/hip_runtime.h>

// ---------------------------------------------------------------------------
// Types / WMMA helpers (CDNA5 gfx1250, wave32)
// ---------------------------------------------------------------------------
typedef __attribute__((ext_vector_type(16))) __bf16 bf16x16;
typedef __attribute__((ext_vector_type(8)))  float  f32x8;
typedef __attribute__((ext_vector_type(4)))  unsigned int u32x4;

union FragU { u32x4 q[2]; bf16x16 v; };
union HalfU { u32x4 q[2]; unsigned short s[16]; };

__device__ __forceinline__ u32x4 zero4() { u32x4 z = {0u, 0u, 0u, 0u}; return z; }
__device__ __forceinline__ f32x8 zero8() { f32x8 z = {0.f,0.f,0.f,0.f,0.f,0.f,0.f,0.f}; return z; }

__device__ __forceinline__ unsigned short f2bf(float f) {
  unsigned int u = __float_as_uint(f);
  u = u + 0x7FFFu + ((u >> 16) & 1u);          // round-to-nearest-even
  return (unsigned short)(u >> 16);
}

// A-matrix 16x32 bf16 fragment: halves {0..7} at p, {16..23} at p+16 (ISA 7.12.2)
__device__ __forceinline__ bf16x16 ld_afrag(const unsigned short* p) {
  FragU u; u.q[0] = *(const u32x4*)(p); u.q[1] = *(const u32x4*)(p + 16); return u.v;
}
// B-matrix 32x16 bf16 fragment: halves {0..7} at p, {16..23} at p+16 (same K order as A)
__device__ __forceinline__ bf16x16 ld_bfrag(const unsigned short* p) {
  FragU u; u.q[0] = *(const u32x4*)(p); u.q[1] = *(const u32x4*)(p + 16); return u.v;
}
__device__ __forceinline__ bf16x16 zfrag() {
  FragU u; u.q[0] = zero4(); u.q[1] = zero4(); return u.v;
}
__device__ __forceinline__ f32x8 wmma_bf16(bf16x16 a, bf16x16 b, f32x8 c) {
  return __builtin_amdgcn_wmma_f32_16x16x32_bf16(false, a, false, b, (short)0, c, false, false);
}

// ---------------------------------------------------------------------------
// Model constants
// ---------------------------------------------------------------------------
constexpr int CB  = 2;      // batch
constexpr int CL  = 3073;   // tokens (3072 patches + cls)
constexpr int CE  = 768;    // embed
constexpr int CH  = 12;     // heads
constexpr int CDH = 64;     // head dim
constexpr int CF  = 3072;   // mlp hidden
constexpr int CNP = 3072;   // patches per batch
constexpr int CKP = 1024;   // patch vector length 4*16*16
constexpr int CNC = 1000;   // classes

// ---------------------------------------------------------------------------
// Generic bf16 WMMA GEMM:  C(MxN) = A(MxK) * Bt(NxK)^T + bias [+Res] [gelu]
// block = 256 threads = 8 waves -> tile 64(M) x 256(N); wave tile 32 x 64.
// OOB A rows / B cols are CLAMPED to 0 (always-load, branch-free K loop);
// the resulting garbage rows/cols are discarded by the store guards.
// mode: 0 = plain, 2 = exact GELU. Cf (fp32) and/or Cb (bf16) outputs.
// ---------------------------------------------------------------------------
__global__ __launch_bounds__(256) void gemm_bf16_kernel(
    const unsigned short* __restrict__ A, int ldA,
    const unsigned short* __restrict__ Bt, int ldB,
    const float* __restrict__ bias,
    const float* Res, int ldR,
    float* Cf, unsigned short* Cb, int ldC,
    int M, int N, int K, int mode)
{
  const int tid   = threadIdx.x;
  const int lane  = tid & 31;
  const int wave  = tid >> 5;
  const int lhalf = lane >> 4;
  const int l16   = lane & 15;
  const int rowbase = blockIdx.y * 64 + (wave >> 2) * 32;
  const int colbase = blockIdx.x * 256 + (wave & 3) * 64;

  f32x8 acc[2][4];
  for (int s = 0; s < 2; ++s)
    for (int t = 0; t < 4; ++t) acc[s][t] = zero8();

  const unsigned short* ap[2];
  for (int s = 0; s < 2; ++s) {
    int arow = rowbase + s * 16 + l16;
    if (arow >= M) arow = 0;                       // clamp: row discarded at store
    ap[s] = A + (long)arow * ldA + lhalf * 8;
  }
  const unsigned short* bp[4];
  for (int t = 0; t < 4; ++t) {
    int col = colbase + t * 16 + l16;
    if (col >= N) col = 0;                         // clamp: col discarded at store
    bp[t] = Bt + (long)col * ldB + lhalf * 8;
  }

  for (int k0 = 0; k0 < K; k0 += 32) {
    bf16x16 a0 = ld_afrag(ap[0] + k0);
    bf16x16 a1 = ld_afrag(ap[1] + k0);
    for (int t = 0; t < 4; ++t) {
      bf16x16 bfv = ld_bfrag(bp[t] + k0);
      acc[0][t] = wmma_bf16(a0, bfv, acc[0][t]);
      acc[1][t] = wmma_bf16(a1, bfv, acc[1][t]);
    }
  }

  for (int t = 0; t < 4; ++t) {
    int col = colbase + t * 16 + l16;
    if (col >= N) continue;
    float bv = bias ? bias[col] : 0.0f;
    for (int s = 0; s < 2; ++s)
      for (int g = 0; g < 8; ++g) {
        int row = rowbase + s * 16 + g + 8 * lhalf;
        if (row >= M) continue;
        float v = acc[s][t][g] + bv;
        if (Res) v += Res[(long)row * ldR + col];
        if (mode == 2) v = 0.5f * v * (1.0f + erff(v * 0.70710678118654752f));
        if (Cf) Cf[(long)row * ldC + col] = v;
        if (Cb) Cb[(long)row * ldC + col] = f2bf(v);
      }
  }
}

// ---------------------------------------------------------------------------
// Dilated attention, one branch. grid = (6 rowtiles, nseg, B*H), 256 threads.
// Segment = 768 gathered tokens (pos = seg*w + j*r + off_h), flash-style over
// 12 j-chunks of 64. Writes o_all[br][b][L][H][64], lse_all[br][b][L][H].
// Padded positions must contribute ZERO k/v (reference softmaxes over them).
// ---------------------------------------------------------------------------
__global__ __launch_bounds__(256) void attn_branch_kernel(
    const unsigned short* __restrict__ qb,
    const unsigned short* __restrict__ kb,
    const unsigned short* __restrict__ vb,
    float* __restrict__ o_all, float* __restrict__ lse_all,
    int br, int w, int r, int gsz)
{
  __shared__ unsigned short Kt[64 * 72];      // [j][d], stride 72 halves
  __shared__ unsigned short Vt[64 * 72];      // [d][j], stride 72 halves
  __shared__ unsigned short Ps[8 * 16 * 72];  // per-wave P scratch [i][j]

  const int tid   = threadIdx.x;
  const int lane  = tid & 31;
  const int wave  = tid >> 5;
  const int lhalf = lane >> 4;
  const int l16   = lane & 15;
  const int b     = blockIdx.z / CH;
  const int h     = blockIdx.z % CH;
  const int seg   = blockIdx.y;
  const int rowt  = blockIdx.x;
  const int off   = h / gsz;
  const long segbase = (long)seg * w + off;

  // ---- preload Q fragments (16 rows x 64 d, stays in registers) ----
  bf16x16 qa0, qa1;
  {
    long qpos = segbase + (long)(rowt * 128 + wave * 16 + l16) * r;
    if (qpos < CL) {
      const unsigned short* qp = qb + (((long)b * CL + qpos) * CE) + h * CDH + lhalf * 8;
      qa0 = ld_afrag(qp);
      qa1 = ld_afrag(qp + 32);
    } else { qa0 = zfrag(); qa1 = zfrag(); }
  }

  float m_run[8], s_run[8];
  f32x8 o_acc[4];
  for (int g = 0; g < 8; ++g) { m_run[g] = -3.0e38f; s_run[g] = 0.0f; }
  for (int t = 0; t < 4; ++t) o_acc[t] = zero8();

  unsigned short* pw = &Ps[wave * 16 * 72];

  for (int jc = 0; jc < 12; ++jc) {
    __syncthreads();
    {   // cooperative K/V chunk load (64 j x 64 d); V stored transposed
      int j = tid >> 2, dseg = tid & 3;
      long kpos = segbase + (long)(jc * 64 + j) * r;
      HalfU kk, vv;
      if (kpos < CL) {
        const unsigned short* kp = kb + (((long)b * CL + kpos) * CE) + h * CDH + dseg * 16;
        const unsigned short* vp = vb + (((long)b * CL + kpos) * CE) + h * CDH + dseg * 16;
        kk.q[0] = *(const u32x4*)kp;      kk.q[1] = *(const u32x4*)(kp + 8);
        vv.q[0] = *(const u32x4*)vp;      vv.q[1] = *(const u32x4*)(vp + 8);
      } else {
        kk.q[0] = zero4(); kk.q[1] = zero4();
        vv.q[0] = zero4(); vv.q[1] = zero4();
      }
      *(u32x4*)&Kt[j * 72 + dseg * 16]     = kk.q[0];
      *(u32x4*)&Kt[j * 72 + dseg * 16 + 8] = kk.q[1];
      for (int e = 0; e < 16; ++e) Vt[(dseg * 16 + e) * 72 + j] = vv.s[e];
    }
    __syncthreads();

    // ---- S = Q K^T (16 x 64), fp32 accum ----
    f32x8 s_acc[4];
    for (int t = 0; t < 4; ++t) {
      const unsigned short* kc = &Kt[(t * 16 + l16) * 72 + lhalf * 8];
      f32x8 a = zero8();
      a = wmma_bf16(qa0, ld_bfrag(kc), a);
      a = wmma_bf16(qa1, ld_bfrag(kc + 32), a);
      s_acc[t] = a;
    }

    // ---- online softmax (row stats reduced over the 16-lane column group) ----
    float alpha[8];
    for (int g = 0; g < 8; ++g) {
      float cm = -3.0e38f;
      for (int t = 0; t < 4; ++t) {
        s_acc[t][g] *= 0.125f;                // Dh^-0.5
        cm = fmaxf(cm, s_acc[t][g]);
      }
      cm = fmaxf(cm, __shfl_xor(cm, 1));
      cm = fmaxf(cm, __shfl_xor(cm, 2));
      cm = fmaxf(cm, __shfl_xor(cm, 4));
      cm = fmaxf(cm, __shfl_xor(cm, 8));
      float nm = fmaxf(m_run[g], cm);
      alpha[g] = __expf(m_run[g] - nm);
      m_run[g] = nm;
      float rs = 0.0f;
      for (int t = 0; t < 4; ++t) {
        float p = __expf(s_acc[t][g] - nm);
        s_acc[t][g] = p;
        rs += p;
      }
      rs += __shfl_xor(rs, 1);
      rs += __shfl_xor(rs, 2);
      rs += __shfl_xor(rs, 4);
      rs += __shfl_xor(rs, 8);
      s_run[g] = s_run[g] * alpha[g] + rs;
    }

    // rescale O, spill P (bf16) to per-wave LDS scratch for re-fragmentation
    for (int t = 0; t < 4; ++t)
      for (int g = 0; g < 8; ++g) {
        o_acc[t][g] *= alpha[g];
        pw[(g + 8 * lhalf) * 72 + t * 16 + l16] = f2bf(s_acc[t][g]);
      }
    __syncthreads();

    // ---- O += P V (K = 64 j) ----
    bf16x16 pa0 = ld_afrag(pw + l16 * 72 + lhalf * 8);
    bf16x16 pa1 = ld_afrag(pw + l16 * 72 + lhalf * 8 + 32);
    for (int t = 0; t < 4; ++t) {
      const unsigned short* vc = &Vt[(t * 16 + l16) * 72 + lhalf * 8];
      o_acc[t] = wmma_bf16(pa0, ld_bfrag(vc), o_acc[t]);
      o_acc[t] = wmma_bf16(pa1, ld_bfrag(vc + 32), o_acc[t]);
    }
  }

  // ---- scatter back to dense token positions ----
  for (int g = 0; g < 8; ++g) {
    int  rowi = rowt * 128 + wave * 16 + g + 8 * lhalf;
    long pos  = segbase + (long)rowi * r;
    if (pos >= CL) continue;
    float inv = 1.0f / s_run[g];
    long obase = ((((long)br * CB + b) * CL + pos) * CH + h) * (long)CDH;
    for (int t = 0; t < 4; ++t)
      o_all[obase + t * 16 + l16] = o_acc[t][g] * inv;
    if (l16 == 0)
      lse_all[(((long)br * CB + b) * CL + pos) * CH + h] = m_run[g] + __logf(s_run[g]);
  }
}

// ---------------------------------------------------------------------------
// LayerNorm: one row (E=768) per 256-thread block -> bf16 output
// ---------------------------------------------------------------------------
__global__ __launch_bounds__(256) void layernorm_kernel(
    const float* __restrict__ x, const float* __restrict__ g,
    const float* __restrict__ bt, unsigned short* __restrict__ out)
{
  __shared__ float red[256];
  long row = blockIdx.x;
  const float* xr = x + row * CE;
  unsigned short* orow = out + row * CE;
  int tid = threadIdx.x;
  float s = 0.f;
  for (int i = tid; i < CE; i += 256) s += xr[i];
  red[tid] = s; __syncthreads();
  for (int st = 128; st > 0; st >>= 1) { if (tid < st) red[tid] += red[tid + st]; __syncthreads(); }
  float mu = red[0] / CE;
  __syncthreads();
  float v = 0.f;
  for (int i = tid; i < CE; i += 256) { float d = xr[i] - mu; v += d * d; }
  red[tid] = v; __syncthreads();
  for (int st = 128; st > 0; st >>= 1) { if (tid < st) red[tid] += red[tid + st]; __syncthreads(); }
  float rstd = rsqrtf(red[0] / CE + 1e-5f);
  for (int i = tid; i < CE; i += 256)
    orow[i] = f2bf((xr[i] - mu) * rstd * g[i] + bt[i]);
}

// ---------------------------------------------------------------------------
// Small elementwise kernels
// ---------------------------------------------------------------------------
__global__ void transpose_conv_kernel(const float* __restrict__ in,
                                      unsigned short* __restrict__ out, int K, int N) {
  long idx = (long)blockIdx.x * 256 + threadIdx.x;
  if (idx >= (long)N * K) return;
  long n = idx / K, k = idx % K;
  out[idx] = f2bf(in[k * (long)N + n]);
}

__global__ void convert_kernel(const float* __restrict__ in,
                               unsigned short* __restrict__ out, long n) {
  long idx = (long)blockIdx.x * 256 + threadIdx.x;
  if (idx < n) out[idx] = f2bf(in[idx]);
}

__global__ void fill_kernel(float* __restrict__ p, float v, long n) {
  long idx = (long)blockIdx.x * 256 + threadIdx.x;
  if (idx < n) p[idx] = v;
}

__global__ void im2col_kernel(const float* __restrict__ x, unsigned short* __restrict__ out) {
  long idx = (long)blockIdx.x * 256 + threadIdx.x;
  if (idx >= (long)CB * CNP * CKP) return;
  long row = idx / CKP;  int kk = (int)(idx % CKP);
  int b  = (int)(row / CNP);
  int p_ = (int)(row % CNP);
  int xd = p_ / 256, rem = p_ % 256, yh = rem / 16, zw = rem % 16;
  int pp = kk / 256, qq = (kk % 256) / 16, rr = kk % 16;
  long src = ((long)b * 48 + (xd * 4 + pp)) * 65536 + (yh * 16 + qq) * 256 + (zw * 16 + rr);
  out[idx] = f2bf(x[src]);
}

__global__ void init_h_kernel(const float* __restrict__ cls, const float* __restrict__ pos,
                              float* __restrict__ h) {
  long idx = (long)blockIdx.x * 256 + threadIdx.x;
  if (idx >= (long)CB * CL * CE) return;
  long rem = idx % ((long)CL * CE);
  int  p   = (int)(rem / CE);
  int  e   = (int)(rem % CE);
  h[idx] = (p == 0) ? (cls[e] + pos[e]) : pos[(long)p * CE + e];
}

// LongNet branch recombination: softmax over branch lse, weighted sum of o.
__global__ void combine_kernel(const float* __restrict__ o_all,
                               const float* __restrict__ lse_all,
                               unsigned short* __restrict__ a_bf) {
  long idx = (long)blockIdx.x * 256 + threadIdx.x;
  if (idx >= (long)CB * CL * CH * CDH) return;
  int  d   = (int)(idx & 63);
  long tmp = idx >> 6;
  int  h   = (int)(tmp % CH);
  long t2  = tmp / CH;
  int  p   = (int)(t2 % CL);
  int  b   = (int)(t2 / CL);
  long base   = ((long)b * CL + p) * CH + h;
  long stride = (long)CB * CL * CH;
  float ls[5], m = -3.0e38f;
  for (int br = 0; br < 5; ++br) { ls[br] = lse_all[br * stride + base]; m = fmaxf(m, ls[br]); }
  float wsum = 0.f, acc = 0.f;
  for (int br = 0; br < 5; ++br) {
    float wt = __expf(ls[br] - m);
    wsum += wt;
    acc  += wt * o_all[(br * stride + base) * CDH + d];
  }
  a_bf[(((long)b * CL + p) * CE) + h * CDH + d] = f2bf(acc / wsum);
}

// ---------------------------------------------------------------------------
// Host orchestration
// ---------------------------------------------------------------------------
static inline long cdiv(long a, long b) { return (a + b - 1) / b; }

extern "C" void kernel_launch(void* const* d_in, const int* in_sizes, int n_in,
                              void* d_out, int out_size, void* d_ws, size_t ws_size,
                              hipStream_t stream) {
  (void)in_sizes; (void)n_in; (void)out_size; (void)ws_size;
  const float* x_in    = (const float*)d_in[0];
  const float* patch_w = (const float*)d_in[1];
  const float* patch_b = (const float*)d_in[2];
  const float* cls_t   = (const float*)d_in[3];
  const float* pos_e   = (const float*)d_in[4];
  const float* ln1_g   = (const float*)d_in[5];
  const float* ln1_b   = (const float*)d_in[6];
  const float* wq      = (const float*)d_in[7];
  const float* bq      = (const float*)d_in[8];
  const float* wk      = (const float*)d_in[9];
  const float* bk      = (const float*)d_in[10];
  const float* wv      = (const float*)d_in[11];
  const float* bv      = (const float*)d_in[12];
  const float* wo      = (const float*)d_in[13];
  const float* bo      = (const float*)d_in[14];
  const float* ln2_g   = (const float*)d_in[15];
  const float* ln2_b   = (const float*)d_in[16];
  const float* w1      = (const float*)d_in[17];
  const float* b1      = (const float*)d_in[18];
  const float* w2      = (const float*)d_in[19];
  const float* b2      = (const float*)d_in[20];
  const float* nf_g    = (const float*)d_in[21];
  const float* nf_b    = (const float*)d_in[22];
  const float* head_w  = (const float*)d_in[23];
  const float* head_b  = (const float*)d_in[24];

  const long BL = (long)CB * CL;

  // bump allocator over workspace
  char*  wsp = (char*)d_ws;
  size_t cur = 0;
  auto alloc = [&](size_t bytes) -> void* {
    cur = (cur + 255) & ~(size_t)255;
    void* p = wsp + cur;
    cur += bytes;
    return p;
  };

  float*          h_buf  = (float*)alloc(BL * CE * 4);
  unsigned short* hn     = (unsigned short*)alloc(BL * CE * 2);
  unsigned short* qbuf   = (unsigned short*)alloc(BL * CE * 2);
  unsigned short* kbuf   = (unsigned short*)alloc(BL * CE * 2);
  unsigned short* vbuf   = (unsigned short*)alloc(BL * CE * 2);
  unsigned short* abuf   = (unsigned short*)alloc(BL * CE * 2);
  unsigned short* ubuf   = (unsigned short*)alloc(BL * CF * 2);
  float*          o_all  = (float*)alloc(5L * BL * CH * CDH * 4);
  float*          lse_all= (float*)alloc(5L * BL * CH * 4);
  unsigned short* pcol   = (unsigned short*)alloc((long)CB * CNP * CKP * 2);
  unsigned short* patchT = (unsigned short*)alloc((long)CE * CKP * 2);
  unsigned short* headT  = (unsigned short*)alloc((long)CNC * CE * 2);
  unsigned short* wqT[2], *wkT[2], *wvT[2], *woT[2], *w1T[2], *w2T[2];
  for (int i = 0; i < 2; ++i) {
    wqT[i] = (unsigned short*)alloc((long)CE * CE * 2);
    wkT[i] = (unsigned short*)alloc((long)CE * CE * 2);
    wvT[i] = (unsigned short*)alloc((long)CE * CE * 2);
    woT[i] = (unsigned short*)alloc((long)CE * CE * 2);
    w1T[i] = (unsigned short*)alloc((long)CF * CE * 2);
    w2T[i] = (unsigned short*)alloc((long)CE * CF * 2);
  }

  auto gemm = [&](const unsigned short* A, int ldA, const unsigned short* Bt, int ldB,
                  const float* bias, const float* Res, int ldR,
                  float* Cf, unsigned short* Cb, int ldC, int M, int N, int K, int mode) {
    dim3 g((unsigned)cdiv(N, 256), (unsigned)cdiv(M, 64));
    gemm_bf16_kernel<<<g, 256, 0, stream>>>(A, ldA, Bt, ldB, bias, Res, ldR, Cf, Cb, ldC,
                                            M, N, K, mode);
  };

  // ---- weight prep (bf16, pre-transposed to [N][K]) ----
  for (int i = 0; i < 2; ++i) {
    long ee = (long)CE * CE;
    transpose_conv_kernel<<<(unsigned)cdiv(ee, 256), 256, 0, stream>>>(wq + i * ee, wqT[i], CE, CE);
    transpose_conv_kernel<<<(unsigned)cdiv(ee, 256), 256, 0, stream>>>(wk + i * ee, wkT[i], CE, CE);
    transpose_conv_kernel<<<(unsigned)cdiv(ee, 256), 256, 0, stream>>>(wv + i * ee, wvT[i], CE, CE);
    transpose_conv_kernel<<<(unsigned)cdiv(ee, 256), 256, 0, stream>>>(wo + i * ee, woT[i], CE, CE);
    long ef = (long)CE * CF;
    transpose_conv_kernel<<<(unsigned)cdiv(ef, 256), 256, 0, stream>>>(w1 + i * ef, w1T[i], CE, CF);
    transpose_conv_kernel<<<(unsigned)cdiv(ef, 256), 256, 0, stream>>>(w2 + i * ef, w2T[i], CF, CE);
  }
  transpose_conv_kernel<<<(unsigned)cdiv((long)CE * CNC, 256), 256, 0, stream>>>(head_w, headT, CE, CNC);
  convert_kernel<<<(unsigned)cdiv((long)CE * CKP, 256), 256, 0, stream>>>(patch_w, patchT, (long)CE * CKP);

  // ---- patch embedding ----
  im2col_kernel<<<(unsigned)cdiv((long)CB * CNP * CKP, 256), 256, 0, stream>>>(x_in, pcol);
  init_h_kernel<<<(unsigned)cdiv(BL * CE, 256), 256, 0, stream>>>(cls_t, pos_e, h_buf);
  for (int b = 0; b < CB; ++b) {
    float* hrow = h_buf + ((long)b * CL + 1) * CE;   // rows 1..3072 of batch b
    gemm(pcol + (long)b * CNP * CKP, CKP, patchT, CKP, patch_b,
         hrow, CE, hrow, nullptr, CE, CNP, CE, CKP, 0);
  }

  // ---- transformer layers ----
  const int bw[5]   = {768, 1536, 3072, 6144, 12288};
  const int brr[5]  = {1, 2, 4, 8, 16};
  const int bgsz[5] = {12, 6, 3, 2, 1};               // ceil(H/r)
  int bseg[5];
  for (int i = 0; i < 5; ++i) bseg[i] = (CL + bw[i] - 1) / bw[i];

  for (int i = 0; i < 2; ++i) {
    layernorm_kernel<<<(unsigned)BL, 256, 0, stream>>>(h_buf, ln1_g + i * CE, ln1_b + i * CE, hn);
    gemm(hn, CE, wqT[i], CE, bq + i * CE, nullptr, 0, nullptr, qbuf, CE, (int)BL, CE, CE, 0);
    gemm(hn, CE, wkT[i], CE, bk + i * CE, nullptr, 0, nullptr, kbuf, CE, (int)BL, CE, CE, 0);
    gemm(hn, CE, wvT[i], CE, bv + i * CE, nullptr, 0, nullptr, vbuf, CE, (int)BL, CE, CE, 0);

    fill_kernel<<<(unsigned)cdiv(5L * BL * CH, 256), 256, 0, stream>>>(lse_all, -1e9f, 5L * BL * CH);
    for (int br = 0; br < 5; ++br) {
      dim3 g(6, (unsigned)bseg[br], (unsigned)(CB * CH));
      attn_branch_kernel<<<g, 256, 0, stream>>>(qbuf, kbuf, vbuf, o_all, lse_all,
                                                br, bw[br], brr[br], bgsz[br]);
    }
    combine_kernel<<<(unsigned)cdiv(BL * CH * CDH, 256), 256, 0, stream>>>(o_all, lse_all, abuf);

    gemm(abuf, CE, woT[i], CE, bo + i * CE, h_buf, CE, h_buf, nullptr, CE, (int)BL, CE, CE, 0);

    layernorm_kernel<<<(unsigned)BL, 256, 0, stream>>>(h_buf, ln2_g + i * CE, ln2_b + i * CE, hn);
    gemm(hn, CE, w1T[i], CE, b1 + i * CF, nullptr, 0, nullptr, ubuf, CF, (int)BL, CF, CE, 2);
    gemm(ubuf, CF, w2T[i], CF, b2 + i * CE, h_buf, CE, h_buf, nullptr, CE, (int)BL, CE, CF, 0);
  }

  // ---- final norm + classification head (cls rows via row stride L*E) ----
  layernorm_kernel<<<(unsigned)BL, 256, 0, stream>>>(h_buf, nf_g, nf_b, hn);
  gemm(hn, CL * CE, headT, CE, head_b, nullptr, 0, (float*)d_out, nullptr, CNC, CB, CNC, CE, 0);
}
